// Net_4612794876089
// MI455X (gfx1250) — matrix operand order repaired and checked by reading, not compiled, not run on hardware
//
#include <hip/hip_runtime.h>
#include <hip/hip_bf16.h>

typedef _Float16 f16;
typedef __attribute__((ext_vector_type(8)))  f16   h8;
typedef __attribute__((ext_vector_type(16))) f16   v16h;
typedef __attribute__((ext_vector_type(8)))  float v8f;

constexpr int B_ = 32, K_ = 32, F_ = 2048, HID_ = 2, NCLS_ = 47;
constexpr int NPAD_ = 48;                 // classifier N padded for WMMA tiles
constexpr float BN_EPS_ = 1e-5f;

// ---------------------------------------------------------------------------
// 1) ysum[b,f] = sum_k neighbor[b,k,0,f]
// ---------------------------------------------------------------------------
__global__ void ysum_kernel(const float* __restrict__ nb, float* __restrict__ ysum) {
    int i = blockIdx.x * blockDim.x + threadIdx.x;     // over B*F
    int b = i / F_, f = i % F_;
    const float* p = nb + (size_t)b * K_ * F_ + f;
    float s = 0.f;
#pragma unroll
    for (int k = 0; k < K_; ++k) s += p[(size_t)k * F_];
    ysum[i] = s;
}

// ---------------------------------------------------------------------------
// 2) denom[b,g] = 1e-7 + sum_f sqrt(|x_f*y_g + x_g*y_f|)   (column normalizer)
//    (x,ysum) interleaved as float2 in LDS -> one ds_load_b64 per iteration,
//    wave-uniform address (broadcast, conflict-free).
// ---------------------------------------------------------------------------
__global__ void denom_kernel(const float* __restrict__ x, const float* __restrict__ ysum,
                             float* __restrict__ denom) {
    __shared__ float2 xy[F_];                          // 16 KB
    int b = blockIdx.y;
    int g = blockIdx.x * 256 + threadIdx.x;
    for (int j = threadIdx.x; j < F_; j += 256)
        xy[j] = make_float2(x[b * F_ + j], ysum[b * F_ + j]);
    __syncthreads();
    float xg = xy[g].x, yg = xy[g].y;
    float acc = 0.f;
#pragma unroll 8
    for (int f = 0; f < F_; ++f) {
        float2 v = xy[f];
        float raw = fmaf(v.x, yg, xg * v.y);
        acc += sqrtf(fabsf(raw));
    }
    denom[b * F_ + g] = acc + 1e-7f;
}

// ---------------------------------------------------------------------------
// 3a) hd[b,c,g] = (w1[c]*x[b,g] + b1[c]) / denom[b,g]      (layer-1 input)
// ---------------------------------------------------------------------------
__global__ void hd1_kernel(const float* __restrict__ x, const float* __restrict__ denom,
                           const float* __restrict__ w1, const float* __restrict__ b1,
                           float* __restrict__ hd) {
    int i = blockIdx.x * blockDim.x + threadIdx.x;     // over B*F
    int b = i / F_, g = i % F_;
    float xv = x[i];
    float inv = 1.0f / denom[i];
    hd[(b * 2 + 0) * F_ + g] = fmaf(w1[0], xv, b1[0]) * inv;
    hd[(b * 2 + 1) * F_ + g] = fmaf(w1[1], xv, b1[1]) * inv;
}

// 3b) layer-2 input: hd[b,c,g] = (sum_i w2[c,i]*hn[b,i,g] + b2[c]) / denom[b,g]
__global__ void hd2_kernel(const float* __restrict__ hn, const float* __restrict__ denom,
                           const float* __restrict__ w2, const float* __restrict__ b2,
                           float* __restrict__ hd) {
    int i = blockIdx.x * blockDim.x + threadIdx.x;
    int b = i / F_, g = i % F_;
    float h0 = hn[(b * 2 + 0) * F_ + g];
    float h1 = hn[(b * 2 + 1) * F_ + g];
    float inv = 1.0f / denom[i];
    hd[(b * 2 + 0) * F_ + g] = (fmaf(w2[0], h0, w2[1] * h1) + b2[0]) * inv;
    hd[(b * 2 + 1) * F_ + g] = (fmaf(w2[2], h0, w2[3] * h1) + b2[1]) * inv;
}

// ---------------------------------------------------------------------------
// 4) Fused featbrd: out[b,c,f] = sum_g sign(raw)*sqrt(|raw|) * hd[b,c,g]
//    fadj is NEVER materialized (saves 512 MB + 1.5 GB of HBM traffic).
//    Two interleaved float2 LDS arrays (32 KB of the 320 KB WGP pool) ->
//    2x ds_load_b64 per inner iteration vs 4x b32; ~6 VALU + 1 v_sqrt_f32
//    per element, transcendental/VALU bound as intended.
// ---------------------------------------------------------------------------
__global__ void layer_kernel(const float* __restrict__ x, const float* __restrict__ ysum,
                             const float* __restrict__ hd, float* __restrict__ hout) {
    __shared__ float2 xy[F_];                          // (x, ysum)      16 KB
    __shared__ float2 hh[F_];                          // (hd0, hd1)     16 KB
    int b = blockIdx.y;
    int f = blockIdx.x * 256 + threadIdx.x;
    for (int j = threadIdx.x; j < F_; j += 256) {
        xy[j] = make_float2(x[b * F_ + j], ysum[b * F_ + j]);
        hh[j] = make_float2(hd[(b * 2 + 0) * F_ + j], hd[(b * 2 + 1) * F_ + j]);
    }
    __syncthreads();
    float xf = xy[f].x, yf = xy[f].y;
    float a0 = 0.f, a1 = 0.f;
#pragma unroll 8
    for (int g = 0; g < F_; ++g) {
        float2 v = xy[g];
        float2 h = hh[g];
        float raw = fmaf(v.x, yf, xf * v.y);           // x_f*y_g + x_g*y_f
        float t = copysignf(sqrtf(fabsf(raw)), raw);   // sgnroot
        a0 = fmaf(t, h.x, a0);
        a1 = fmaf(t, h.y, a1);
    }
    hout[(b * 2 + 0) * F_ + f] = a0;
    hout[(b * 2 + 1) * F_ + f] = a1;
}

// ---------------------------------------------------------------------------
// 5) BatchNorm stats per channel over (B,F): one block per channel
// ---------------------------------------------------------------------------
__global__ void bn_stats_kernel(const float* __restrict__ h, float* __restrict__ stats) {
    __shared__ float ssum[256];
    __shared__ float ssq[256];
    int c = blockIdx.x;
    float s = 0.f, q = 0.f;
    for (int i = threadIdx.x; i < B_ * F_; i += 256) {
        int b = i / F_, f = i % F_;
        float v = h[(b * 2 + c) * F_ + f];
        s += v;
        q = fmaf(v, v, q);
    }
    ssum[threadIdx.x] = s;
    ssq[threadIdx.x] = q;
    __syncthreads();
    for (int off = 128; off > 0; off >>= 1) {
        if (threadIdx.x < off) {
            ssum[threadIdx.x] += ssum[threadIdx.x + off];
            ssq[threadIdx.x]  += ssq[threadIdx.x + off];
        }
        __syncthreads();
    }
    if (threadIdx.x == 0) {
        stats[c * 2 + 0] = ssum[0];
        stats[c * 2 + 1] = ssq[0];
    }
}

// 6) normalize + softsign; optionally emit f16 for the WMMA classifier
__global__ void bn_softsign_kernel(const float* __restrict__ h, const float* __restrict__ stats,
                                   const float* __restrict__ gamma, const float* __restrict__ beta,
                                   float* __restrict__ hn, f16* __restrict__ flat16) {
    int i = blockIdx.x * blockDim.x + threadIdx.x;     // over B*2*F
    int c = (i / F_) & 1;
    float n = (float)(B_ * F_);
    float mean = stats[c * 2 + 0] / n;
    float var  = stats[c * 2 + 1] / n - mean * mean;
    float y = fmaf(gamma[c] * (h[i] - mean), rsqrtf(var + BN_EPS_), beta[c]);
    float sy = y / (1.0f + fabsf(y));
    hn[i] = sy;
    if (flat16) flat16[i] = (f16)sy;
}

// 7) Wc -> f16, padded to 48 rows (row >= 47 zeroed so WMMA needs no masking)
__global__ void wc16_kernel(const float* __restrict__ Wc, f16* __restrict__ wc16) {
    int i = blockIdx.x * blockDim.x + threadIdx.x;     // over NPAD*HID*F
    int row = i / (HID_ * F_);
    wc16[i] = (row < NCLS_) ? (f16)Wc[i] : (f16)0.f;
}

// ---------------------------------------------------------------------------
// 8) Classifier GEMM via V_WMMA_F32_16X16X32_F16.
//    One wave (32 threads) per 16x16 output tile; M=32 (2 tiles), N=48 (3),
//    K=4096 (128 WMMA steps). Register layouts per CDNA5 ISA:
//      A (16x32 f16): halves 0..7 -> K = hi*8 + t, halves 8..15 -> K = 16+hi*8+t
//      B (32x16 f16): same K order as A,                      N = lane&15
//      D (16x16 f32): VGPR r      -> M = r + hi*8,             N = lane&15
// ---------------------------------------------------------------------------
__global__ void cls_wmma_kernel(const f16* __restrict__ A, const f16* __restrict__ Bm,
                                const float* __restrict__ bc, float* __restrict__ out) {
    int mt = blockIdx.x % 2;           // 2 tiles over B_=32
    int nt = blockIdx.x / 2;           // 3 tiles over NPAD_=48
    int lane = threadIdx.x;            // wave32: one wave per block, EXEC all-1s
    int hi = lane >> 4;
    int lo = lane & 15;
    const int KK = HID_ * F_;          // 4096
    const f16* arow = A  + (size_t)(mt * 16 + lo) * KK;   // A[m, :]
    const f16* brow = Bm + (size_t)(nt * 16 + lo) * KK;   // B[k,n] = Wc[n,k]
    v8f acc = {};
    for (int kb = 0; kb < KK; kb += 32) {
        h8 alo = *(const h8*)(arow + kb + hi * 8);
        h8 ahi = *(const h8*)(arow + kb + 16 + hi * 8);
        h8 blo = *(const h8*)(brow + kb + hi * 8);
        h8 bhi = *(const h8*)(brow + kb + 16 + hi * 8);
        v16h av, bv;
#pragma unroll
        for (int t = 0; t < 8; ++t) {
            av[t] = alo[t]; av[t + 8] = ahi[t];
            bv[t] = blo[t]; bv[t + 8] = bhi[t];
        }
        acc = __builtin_amdgcn_wmma_f32_16x16x32_f16(
            /*neg_a=*/false, av, /*neg_b=*/false, bv,
            /*c_mod=*/(short)0, acc, /*reuse_a=*/false, /*reuse_b=*/false);
    }
    int n = nt * 16 + lo;
    if (n < NCLS_) {
        float bias = bc[n];
#pragma unroll
        for (int r = 0; r < 8; ++r) {
            int m = mt * 16 + r + hi * 8;
            out[m * NCLS_ + n] = acc[r] + bias;
        }
    }
}

// ---------------------------------------------------------------------------
extern "C" void kernel_launch(void* const* d_in, const int* in_sizes, int n_in,
                              void* d_out, int out_size, void* d_ws, size_t ws_size,
                              hipStream_t stream) {
    const float* x   = (const float*)d_in[0];   // (B,1,F)
    const float* nb  = (const float*)d_in[1];   // (B,K,1,F)
    const float* w1  = (const float*)d_in[2];   // (HID,1)
    const float* b1  = (const float*)d_in[3];
    const float* g1  = (const float*)d_in[4];
    const float* be1 = (const float*)d_in[5];
    const float* w2  = (const float*)d_in[6];   // (HID,HID)
    const float* b2  = (const float*)d_in[7];
    const float* g2  = (const float*)d_in[8];
    const float* be2 = (const float*)d_in[9];
    const float* Wc  = (const float*)d_in[10];  // (NCLS, HID*F)
    const float* bc  = (const float*)d_in[11];
    float* out = (float*)d_out;                 // (B, NCLS)

    char* ws = (char*)d_ws;                     // ~2.8 MB used
    float* ysum   = (float*)(ws + 0);           // 256 KB
    float* denom  = (float*)(ws + 262144);      // 256 KB
    float* hd     = (float*)(ws + 524288);      // 512 KB
    float* hout   = (float*)(ws + 1048576);     // 512 KB
    float* hn     = (float*)(ws + 1572864);     // 512 KB
    float* stats  = (float*)(ws + 2097152);     // 128 B
    f16*   flat16 = (f16*)(ws + 2097280);       // 256 KB (16B-aligned)
    f16*   wc16   = (f16*)(ws + 2359424);       // 384 KB (16B-aligned)

    // neighbor reduce
    ysum_kernel<<<(B_ * F_) / 256, 256, 0, stream>>>(nb, ysum);
    // column normalizers of implicit fadj
    denom_kernel<<<dim3(F_ / 256, B_), 256, 0, stream>>>(x, ysum, denom);
    // layer 1
    hd1_kernel<<<(B_ * F_) / 256, 256, 0, stream>>>(x, denom, w1, b1, hd);
    layer_kernel<<<dim3(F_ / 256, B_), 256, 0, stream>>>(x, ysum, hd, hout);
    bn_stats_kernel<<<2, 256, 0, stream>>>(hout, stats);
    bn_softsign_kernel<<<(B_ * 2 * F_) / 256, 256, 0, stream>>>(hout, stats, g1, be1, hn, nullptr);
    // layer 2
    hd2_kernel<<<(B_ * F_) / 256, 256, 0, stream>>>(hn, denom, w2, b2, hd);
    layer_kernel<<<dim3(F_ / 256, B_), 256, 0, stream>>>(x, ysum, hd, hout);
    bn_stats_kernel<<<2, 256, 0, stream>>>(hout, stats);
    bn_softsign_kernel<<<(B_ * 2 * F_) / 256, 256, 0, stream>>>(hout, stats, g2, be2, hn, flat16);
    // classifier (WMMA)
    wc16_kernel<<<(NPAD_ * HID_ * F_) / 256, 256, 0, stream>>>(Wc, wc16);
    cls_wmma_kernel<<<6, 32, 0, stream>>>(flat16, wc16, bc, out);
}
